// GraphProcessor_68693706932423
// MI455X (gfx1250) — matrix operand order repaired and checked by reading, not compiled, not run on hardware
//
#include <hip/hip_runtime.h>
#include <hip/hip_bf16.h>
#include <math.h>

#define NEG_SLOPE 0.2f
#define DHEAD 48

typedef __bf16 v16bf __attribute__((ext_vector_type(16)));
typedef __bf16 v8bf  __attribute__((ext_vector_type(8)));
typedef float  v8f   __attribute__((ext_vector_type(8)));

__device__ __forceinline__ __bf16 f2bf(float f) {
  union { float f; unsigned u; } v; v.f = f;
  unsigned r = v.u + 0x7FFFu + ((v.u >> 16) & 1u);   // round-to-nearest-even
  unsigned short s = (unsigned short)(r >> 16);
  union { unsigned short s; __bf16 b; } o; o.s = s;
  return o.b;
}

// ---------------- fp32 -> bf16 staging ----------------
__global__ void cvt_f32_bf16(const float* __restrict__ in, __bf16* __restrict__ out, int n) {
  int i = blockIdx.x * blockDim.x + threadIdx.x;
  int stride = gridDim.x * blockDim.x;
  for (; i < n; i += stride) out[i] = f2bf(in[i]);
}

// ---------------- pack W (fp32 [K,Nc]) into WMMA B-fragment layout, fused cvt ----------------
// Bp element index: ((ks*tilesN + tn)*32 + lane)*16 + e
//   maps to B[k, c] with k = ks*32 + (lane>=16 ? 8:0) + (e&7) + (e>=8 ? 16:0),  c = tn*16 + lane%16
__global__ void pack_b_kernel(const float* __restrict__ W, __bf16* __restrict__ Bp,
                              int K, int Nc) {
  int i = blockIdx.x * blockDim.x + threadIdx.x;
  if (i >= K * Nc) return;
  int e    = i & 15;
  int lane = (i >> 4) & 31;
  int tile = i >> 9;                 // ks*tilesN + tn
  int tilesN = Nc >> 4;
  int ks = tile / tilesN;
  int tn = tile % tilesN;
  int k = ks * 32 + ((lane >= 16) ? 8 : 0) + (e & 7) + ((e >> 3) << 4);
  int c = tn * 16 + (lane & 15);
  Bp[i] = f2bf(W[(size_t)k * Nc + c]);
}

// ---------------- WMMA GEMM: C[M,Nc] = A[M,K] @ B[K,Nc] ----------------
// one wave per 16x(16*NT) output tile strip; B pre-packed in fragment layout
template <int NT>
__global__ void gemm_bf16_wmma(const __bf16* __restrict__ A, const __bf16* __restrict__ Bp,
                               float* __restrict__ C, int M, int K, int Nc) {
  int tilesN  = Nc >> 4;
  int groupsN = tilesN / NT;
  int tilesM  = (M + 15) >> 4;
  int wave = (blockIdx.x * blockDim.x + threadIdx.x) >> 5;
  int lane = threadIdx.x & 31;
  if (wave >= tilesM * groupsN) return;        // wave-uniform: EXEC stays all-ones
  int tm = wave / groupsN;
  int gn = wave % groupsN;
  int l16 = lane & 15;
  bool hi = lane >= 16;

  int row = tm * 16 + l16;
  if (row >= M) row = M - 1;                   // safety (M is 16-aligned here)
  const __bf16* arow = A + (size_t)row * K + (hi ? 8 : 0);

  v8f acc[NT];
#pragma unroll
  for (int t = 0; t < NT; t++) acc[t] = (v8f){};

  int ksteps = K >> 5;
  for (int ks = 0; ks < ksteps; ks++) {
    // A fragment: two contiguous 16B loads (ISA 7.12.2 16-bit A 16x32 layout)
    v8bf alo = *(const v8bf*)(arow + ks * 32);
    v8bf ahi = *(const v8bf*)(arow + ks * 32 + 16);
    v16bf a;
#pragma unroll
    for (int i = 0; i < 8; i++) { a[i] = alo[i]; a[8 + i] = ahi[i]; }
#pragma unroll
    for (int t = 0; t < NT; t++) {
      int tn = gn * NT + t;
      const v16bf* bf =
          (const v16bf*)(Bp + (((size_t)ks * tilesN + tn) * 32 + lane) * 16);
      acc[t] = __builtin_amdgcn_wmma_f32_16x16x32_bf16(false, a, false, *bf, (short)0,
                                                       acc[t], false, false);
    }
  }
  // C/D layout: VGPR r -> M = tm*16 + r + (hi ? 8 : 0)
  int rbase = tm * 16 + (hi ? 8 : 0);
#pragma unroll
  for (int t = 0; t < NT; t++) {
    int col = (gn * NT + t) * 16 + l16;
#pragma unroll
    for (int r = 0; r < 8; r++) {
      int orow = rbase + r;
      if (orow < M) C[(size_t)orow * Nc + col] = acc[t][r];
    }
  }
}

// ---------------- attention logits: a = sum_d h[n,h,d]*att[h,d] ----------------
__global__ void attn_kernel(const float* __restrict__ h, const float* __restrict__ att_s,
                            const float* __restrict__ att_d, float* __restrict__ a_src,
                            float* __restrict__ a_dst, int N, int H) {
  int idx = blockIdx.x * blockDim.x + threadIdx.x;
  if (idx >= N * H) return;
  int n = idx / H, hh = idx % H;
  const float* hp = h + (size_t)n * H * DHEAD + hh * DHEAD;
  const float* as = att_s + hh * DHEAD;
  const float* ad = att_d + hh * DHEAD;
  float ss = 0.f, sd = 0.f;
#pragma unroll 8
  for (int d = 0; d < DHEAD; d++) { float v = hp[d]; ss += v * as[d]; sd += v * ad[d]; }
  a_src[idx] = ss; a_dst[idx] = sd;
}

// ---------------- init m/denom/acc ----------------
__global__ void init_kernel(float* __restrict__ m, float* __restrict__ denom,
                            float* __restrict__ acc, int NH, int NHD) {
  int i = blockIdx.x * blockDim.x + threadIdx.x;
  int stride = gridDim.x * blockDim.x;
  for (int j = i; j < NHD; j += stride) acc[j] = 0.f;
  for (int j = i; j < NH; j += stride) { m[j] = -3.402823e38f; denom[j] = 0.f; }
}

__device__ __forceinline__ void atomicMaxF(float* addr, float val) {
  if (val >= 0.f) atomicMax((int*)addr, __float_as_int(val));
  else            atomicMin((unsigned int*)addr, (unsigned int)__float_as_int(val));
}

// ---------------- edge pass 1: segment max of leaky(e) ----------------
__global__ void edge_max_kernel(const int* __restrict__ ei, int E, int N,
                                const float* __restrict__ a_src, const float* __restrict__ a_dst,
                                float* __restrict__ m, int H) {
  int e = blockIdx.x * blockDim.x + threadIdx.x;
  int ET = E + N;
  if (e >= ET) return;
  int s, d;
  if (e < E) { s = ei[e]; d = ei[(size_t)E + e]; } else { s = d = e - E; }
  for (int hh = 0; hh < H; hh++) {
    float v = a_src[(size_t)s * H + hh] + a_dst[(size_t)d * H + hh];
    v = v > 0.f ? v : NEG_SLOPE * v;
    atomicMaxF(&m[(size_t)d * H + hh], v);
  }
}

// ---------------- edge pass 2: acc[dst] += h[src]*exp(e-m), denom[dst] += exp ----------------
// one wave32 per edge: coalesced 128B gather/scatter per iteration
__global__ void edge_msg_kernel(const int* __restrict__ ei, int E, int N,
                                const float* __restrict__ h, const float* __restrict__ a_src,
                                const float* __restrict__ a_dst, const float* __restrict__ m,
                                float* __restrict__ denom, float* __restrict__ acc,
                                int H, int HD) {
  int wid = (blockIdx.x * blockDim.x + threadIdx.x) >> 5;
  int lane = threadIdx.x & 31;
  int ET = E + N;
  if (wid >= ET) return;                         // wave-uniform exit
  int s, d;
  if (wid < E) { s = ei[wid]; d = ei[(size_t)E + wid]; } else { s = d = wid - E; }
  float eh[4];
  for (int hh = 0; hh < H; hh++) {
    float v = a_src[(size_t)s * H + hh] + a_dst[(size_t)d * H + hh];
    v = v > 0.f ? v : NEG_SLOPE * v;
    eh[hh] = __expf(v - m[(size_t)d * H + hh]);
  }
  if (lane < H) atomicAdd(&denom[(size_t)d * H + lane], eh[lane]);
  const float* hs = h + (size_t)s * HD;
  float* ad = acc + (size_t)d * HD;
  for (int c = lane; c < HD; c += 32)
    atomicAdd(&ad[c], hs[c] * eh[c / DHEAD]);
}

// ---------------- finalize: x_next = relu(acc/denom + bias) ----------------
__global__ void finalize_kernel(const float* __restrict__ acc, const float* __restrict__ denom,
                                const float* __restrict__ bias, float* __restrict__ xn,
                                int N, int H, int HD) {
  int i = blockIdx.x * blockDim.x + threadIdx.x;
  if (i >= N * HD) return;
  int n = i / HD, c = i % HD;
  float v = acc[i] / (denom[(size_t)n * H + c / DHEAD] + 1e-16f) + bias[c];
  xn[i] = v > 0.f ? v : 0.f;
}

// ---------------- global mean pool (batch is sorted) ----------------
__device__ __forceinline__ int lower_bound_i(const int* b, int n, int key) {
  int lo = 0, hi = n;
  while (lo < hi) { int mid = (lo + hi) >> 1; if (b[mid] < key) lo = mid + 1; else hi = mid; }
  return lo;
}
__global__ void pool_kernel(const float* __restrict__ x, const int* __restrict__ batch,
                            float* __restrict__ pooled, int N, int C) {
  int g = blockIdx.x, c = threadIdx.x;
  int s = lower_bound_i(batch, N, g);
  int e = lower_bound_i(batch, N, g + 1);
  float sum = 0.f;
  for (int n = s; n < e; n++) sum += x[(size_t)n * C + c];
  pooled[(size_t)g * C + c] = sum / fmaxf((float)(e - s), 1.0f);
}

// ---------------- small dense FC ----------------
__global__ void fc_kernel(const float* __restrict__ in, const float* __restrict__ W,
                          const float* __restrict__ b, float* __restrict__ out,
                          int M, int K, int Nc, int do_relu) {
  int i = blockIdx.x * blockDim.x + threadIdx.x;
  if (i >= M * Nc) return;
  int mi = i / Nc, c = i % Nc;
  float sum = b[c];
  for (int k = 0; k < K; k++) sum += in[(size_t)mi * K + k] * W[(size_t)k * Nc + c];
  if (do_relu) sum = sum > 0.f ? sum : 0.f;
  out[i] = sum;
}

extern "C" void kernel_launch(void* const* d_in, const int* in_sizes, int n_in,
                              void* d_out, int out_size, void* d_ws, size_t ws_size,
                              hipStream_t stream) {
  const float* x      = (const float*)d_in[0];
  const int*   eidx   = (const int*)d_in[1];
  const int*   batch  = (const int*)d_in[2];
  const int N  = in_sizes[0] / 128;
  const int E  = in_sizes[1] / 2;
  const int ET = E + N;

  const float* Wl[3]   = { (const float*)d_in[3],  (const float*)d_in[7],  (const float*)d_in[11] };
  const float* atS[3]  = { (const float*)d_in[4],  (const float*)d_in[8],  (const float*)d_in[12] };
  const float* atD[3]  = { (const float*)d_in[5],  (const float*)d_in[9],  (const float*)d_in[13] };
  const float* bia[3]  = { (const float*)d_in[6],  (const float*)d_in[10], (const float*)d_in[14] };
  const float* fcW1 = (const float*)d_in[15];
  const float* fcb1 = (const float*)d_in[16];
  const float* fcW2 = (const float*)d_in[17];
  const float* fcb2 = (const float*)d_in[18];

  const int HEADS_[3] = {4, 2, 2};
  const int CIN_[3]   = {128, 192, 96};

  // workspace layout (floats first; bf16 region stays 32B-aligned)
  float* hbuf  = (float*)d_ws;                       // N*192
  float* acc   = hbuf  + (size_t)N * 192;            // N*192
  float* a_src = acc   + (size_t)N * 192;            // N*4
  float* a_dst = a_src + (size_t)N * 4;              // N*4
  float* mseg  = a_dst + (size_t)N * 4;              // N*4
  float* denom = mseg  + (size_t)N * 4;              // N*4
  float* pooled= denom + (size_t)N * 4;              // 64*96
  float* fc1o  = pooled + 64 * 96;                   // 64*192
  __bf16* xb   = (__bf16*)(fc1o + 64 * 192);         // N*192 bf16
  __bf16* Bp   = xb + (size_t)N * 192;               // 192*192 bf16 (packed fragments)

  const int BT = 256;
  const float* xin = x;
  for (int l = 0; l < 3; l++) {
    int H  = HEADS_[l];
    int K  = CIN_[l];
    int HD = H * DHEAD;
    // stage A to bf16; pack+convert W into fragment layout
    int na = N * K;
    cvt_f32_bf16<<<dim3((na + BT - 1) / BT), dim3(BT), 0, stream>>>(xin, xb, na);
    int nw = K * HD;
    pack_b_kernel<<<dim3((nw + BT - 1) / BT), dim3(BT), 0, stream>>>(Wl[l], Bp, K, HD);
    // h = x @ W  (WMMA bf16 -> f32), NT=2 column tiles per wave
    int waves = ((N + 15) / 16) * ((HD / 16) / 2);
    gemm_bf16_wmma<2><<<dim3((waves + 7) / 8), dim3(BT), 0, stream>>>(xb, Bp, hbuf, N, K, HD);
    // attention logits
    attn_kernel<<<dim3((N * H + BT - 1) / BT), dim3(BT), 0, stream>>>(hbuf, atS[l], atD[l],
                                                                      a_src, a_dst, N, H);
    // init segment state
    init_kernel<<<dim3((N * HD + BT - 1) / BT), dim3(BT), 0, stream>>>(mseg, denom, acc,
                                                                       N * H, N * HD);
    // segment max
    edge_max_kernel<<<dim3((ET + BT - 1) / BT), dim3(BT), 0, stream>>>(eidx, E, N, a_src,
                                                                       a_dst, mseg, H);
    // weighted message scatter (wave per edge)
    edge_msg_kernel<<<dim3((ET + 7) / 8), dim3(BT), 0, stream>>>(eidx, E, N, hbuf, a_src,
                                                                 a_dst, mseg, denom, acc, H, HD);
    // normalize + bias + relu -> next layer's input (reuses hbuf)
    finalize_kernel<<<dim3((N * HD + BT - 1) / BT), dim3(BT), 0, stream>>>(acc, denom, bia[l],
                                                                           hbuf, N, H, HD);
    xin = hbuf;
  }

  // global mean pool over sorted batch (no atomics)
  pool_kernel<<<dim3(64), dim3(96), 0, stream>>>(hbuf, batch, pooled, N, 96);
  // MLP head
  fc_kernel<<<dim3((64 * 192 + BT - 1) / BT), dim3(BT), 0, stream>>>(pooled, fcW1, fcb1, fc1o,
                                                                     64, 96, 192, 1);
  fc_kernel<<<dim3((64 * 96 + BT - 1) / BT), dim3(BT), 0, stream>>>(fc1o, fcW2, fcb2,
                                                                    (float*)d_out, 64, 192, 96, 0);
}
